// ENCE_10514079941568
// MI455X (gfx1250) — matrix operand order repaired and checked by reading, not compiled, not run on hardware
//
#include <hip/hip_runtime.h>
#include <hip/hip_bf16.h>
#include <math.h>
#include <stdint.h>

namespace {
constexpr int NN   = 50000;   // nodes
constexpr int EE   = 800000;  // edges per graph
constexpr int INF_ = 128;     // input feature dim
constexpr int HID  = 64;      // fc hidden
constexpr int H    = 4;       // heads
constexpr int D    = 64;      // dim per head
constexpr int F    = 256;     // H*D
constexpr int OUTF = 16;
constexpr int SAH  = 128;
constexpr float NEG_SLOPE = 0.2f;
constexpr int LDS_STRIDE = 40;   // ushort elems; 80 B row: 16B-aligned, bank-conflict-free
}

// ---------------- WMMA types (CDNA5 gfx1250, wave32) ----------------
typedef __attribute__((ext_vector_type(16))) __bf16 v16bf;
typedef __attribute__((ext_vector_type(8)))  float  v8f;

union FragBF { v16bf v; unsigned short u[16]; uint4 q[2]; };
union FragF8 { v8f   v; float          f[8];  };

#if defined(__gfx1250__) && __has_builtin(__builtin_amdgcn_global_load_async_to_lds_b128) && __has_builtin(__builtin_amdgcn_s_wait_asynccnt)
#define HAVE_ASYNC_LDS 1
// builtin parameter type per hipcc diagnostic: vector_size(16) int, AS(1)/AS(3)
typedef int v4i_gv __attribute__((__vector_size__(4 * sizeof(int))));
#define AS1P __attribute__((address_space(1))) v4i_gv*
#define AS3P __attribute__((address_space(3))) v4i_gv*
#else
#define HAVE_ASYNC_LDS 0
#endif

__device__ __forceinline__ unsigned short f2bf(float x) {
  unsigned u = __float_as_uint(x);
  u += 0x7FFFu + ((u >> 16) & 1u);   // round-to-nearest-even
  return (unsigned short)(u >> 16);
}

// order-preserving float<->uint encoding for atomicMax on floats
__device__ __forceinline__ unsigned encf(float f) {
  unsigned u = __float_as_uint(f);
  return (u & 0x80000000u) ? ~u : (u | 0x80000000u);
}
__device__ __forceinline__ float decf(unsigned x) {
  return (x & 0x80000000u) ? __uint_as_float(x ^ 0x80000000u)
                           : __uint_as_float(~x);
}

// ---------------- generic bf16-WMMA GEMM: C = act(A*Bt^T + bias) ----------------
// A:  [M,K]  bf16 row-major (lda, elems), pre-converted
// Bt: [Nc,K] bf16 row-major (pre-transposed B) -> both stage as pure 16B copies
// C:  [M,Nc] f32 (ldc)
// block = 256 threads = 8 waves; tile 64(M) x 64(N); K stepped by 32.
// DOUBLE-BUFFERED async global->LDS staging: the next K-step's copies are in
// flight behind the current step's WMMAs (ASYNCcnt decrements in issue order,
// so s_wait_asynccnt 2 waits only for the current tile's two copies).
// Out-of-range rows/cols are index-CLAMPED on load (their products only land in
// C elements we never store) -> branch-free inner loop, EXEC stays all-1s.
template <bool HASBIAS, int ACT>
__global__ __launch_bounds__(256) void gemm_bf16_wmma(
    const unsigned short* __restrict__ A, int lda,
    const unsigned short* __restrict__ Bt,
    float* __restrict__ C, int ldc,
    int M, int Nc, int K,
    const float* __restrict__ bias)
{
  // A staged row-major [row][k]; B staged [n][k] (already transposed in global)
  __shared__ __align__(16) unsigned short AsBuf[2][64 * LDS_STRIDE];
  __shared__ __align__(16) unsigned short BsBuf[2][64 * LDS_STRIDE];

  const int t    = threadIdx.x;
  const int wave = t >> 5;
  const int lane = t & 31;
  const int half = lane >> 4;
  const int l16  = lane & 15;
  const int wm   = wave >> 1;          // 0..3 : 16-row strip
  const int wn   = wave & 1;           // 0..1 : 32-col strip
  const int rowBase = blockIdx.y * 64;
  const int colBase = blockIdx.x * 64;

  // per-thread staging job: one 16B chunk of A tile + one 16B chunk of B tile
  const int r4  = t >> 2;              // 0..63
  const int kq8 = (t & 3) << 3;        // 0,8,16,24
  const int grA = min(rowBase + r4, M - 1);
  const int gcB = min(colBase + r4, Nc - 1);
  const unsigned short* gA = A  + (size_t)grA * lda + kq8;
  const unsigned short* gB = Bt + (size_t)gcB * K   + kq8;
  const int ldsOff = r4 * LDS_STRIDE + kq8;

  auto stage = [&](int step, int pb) {
    const unsigned short* ga = gA + step * 32;
    const unsigned short* gb = gB + step * 32;
    unsigned short* la = &AsBuf[pb][ldsOff];
    unsigned short* lb = &BsBuf[pb][ldsOff];
#if HAVE_ASYNC_LDS
    __builtin_amdgcn_global_load_async_to_lds_b128(
        (AS1P)(uintptr_t)ga, (AS3P)(uintptr_t)la, 0, 0);
    __builtin_amdgcn_global_load_async_to_lds_b128(
        (AS1P)(uintptr_t)gb, (AS3P)(uintptr_t)lb, 0, 0);
#else
    *(uint4*)la = *(const uint4*)ga;
    *(uint4*)lb = *(const uint4*)gb;
#endif
  };

  v8f c0 = {0.f,0.f,0.f,0.f,0.f,0.f,0.f,0.f};
  v8f c1 = {0.f,0.f,0.f,0.f,0.f,0.f,0.f,0.f};

  const int nsteps = K >> 5;           // K is a multiple of 32
  stage(0, 0);
  int p = 0;
  for (int s = 0; s < nsteps; ++s) {
    if (s + 1 < nsteps) {
      stage(s + 1, p ^ 1);             // overlap next tile behind this step
#if HAVE_ASYNC_LDS
      __builtin_amdgcn_s_wait_asynccnt(2);   // only wait for CURRENT tile's 2
#endif
    } else {
#if HAVE_ASYNC_LDS
      __builtin_amdgcn_s_wait_asynccnt(0);
#endif
    }
    __syncthreads();

    // ---- fragments as contiguous b128 LDS loads (ISA 16-bit A/B lane layouts)
    // A lane data: k in [8*half, 8*half+8) and [16+8*half, ...+8)
    FragBF a, b0, b1;
    const unsigned short* ap = &AsBuf[p][(wm * 16 + l16) * LDS_STRIDE + 8 * half];
    a.q[0] = *(const uint4*)(ap);
    a.q[1] = *(const uint4*)(ap + 16);
    // B lane data: k in [8*half, 8*half+8) and [16+8*half, ...+8) at column n
    const unsigned short* bp0 = &BsBuf[p][(wn * 32 + l16) * LDS_STRIDE + 8 * half];
    b0.q[0] = *(const uint4*)(bp0);
    b0.q[1] = *(const uint4*)(bp0 + 16);
    const unsigned short* bp1 = bp0 + 16 * LDS_STRIDE;
    b1.q[0] = *(const uint4*)(bp1);
    b1.q[1] = *(const uint4*)(bp1 + 16);

    c0 = __builtin_amdgcn_wmma_f32_16x16x32_bf16(false, a.v, false, b0.v,
                                                 (short)0, c0, false, false);
    c1 = __builtin_amdgcn_wmma_f32_16x16x32_bf16(false, a.v, false, b1.v,
                                                 (short)0, c1, false, false);
    __syncthreads();
    p ^= 1;
  }

  // epilogue: C/D layout — VGPR i holds M=i (lanes 0-15) / M=i+8 (lanes 16-31)
  FragF8 r0, r1; r0.v = c0; r1.v = c1;
  const int gn0 = colBase + wn * 32 + l16;
  const int gn1 = gn0 + 16;
  float bias0 = 0.f, bias1 = 0.f;
  if (HASBIAS) {
    bias0 = bias[min(gn0, Nc - 1)];
    bias1 = bias[min(gn1, Nc - 1)];
  }
#pragma unroll
  for (int i = 0; i < 8; ++i) {
    int gm = rowBase + wm * 16 + (half << 3) + i;
    if (gm >= M) continue;
    if (gn0 < Nc) {
      float vv = r0.f[i] + bias0;
      if (ACT == 1) vv = tanhf(vv);
      C[(size_t)gm * ldc + gn0] = vv;
    }
    if (gn1 < Nc) {
      float vv = r1.f[i] + bias1;
      if (ACT == 1) vv = tanhf(vv);
      C[(size_t)gm * ldc + gn1] = vv;
    }
  }
}

// ---------------- conversion / transpose kernels ----------------
__global__ void cvt_bf16(const float* __restrict__ in, unsigned short* __restrict__ out, long n) {
  long i = (long)blockIdx.x * blockDim.x + threadIdx.x;
  if (i < n) out[i] = f2bf(in[i]);
}
// Bt[n*K + k] = bf16(B[k*ldb + n])
__global__ void transpose_bf16(const float* __restrict__ B, unsigned short* __restrict__ Bt,
                               int K, int Nc, int ldb) {
  int i = blockIdx.x * blockDim.x + threadIdx.x;
  if (i >= K * Nc) return;
  int n = i / K, k = i - n * K;
  Bt[i] = f2bf(B[(size_t)k * ldb + n]);
}

// ---------------- small utility kernels ----------------
__global__ void zero_f32(float* p, long n) {
  long i = (long)blockIdx.x * blockDim.x + threadIdx.x;
  if (i < n) p[i] = 0.f;
}
__global__ void fill_u32(unsigned* p, unsigned v, long n) {
  long i = (long)blockIdx.x * blockDim.x + threadIdx.x;
  if (i < n) p[i] = v;
}

// el[n,h] = sum_d z[n,h,d]*al[h,d] ; er likewise
__global__ void attn_scores(const float* __restrict__ z,
                            const float* __restrict__ al,
                            const float* __restrict__ ar,
                            float* __restrict__ el, float* __restrict__ er)
{
  int idx = blockIdx.x * blockDim.x + threadIdx.x;   // n*H + h
  if (idx >= NN * H) return;
  int n = idx >> 2, hh = idx & 3;
  const float4* zp = (const float4*)(z + (size_t)n * F + hh * D);
  const float4* ap = (const float4*)(al + hh * D);
  const float4* rp = (const float4*)(ar + hh * D);
  float sl = 0.f, sr = 0.f;
#pragma unroll
  for (int q = 0; q < D / 4; ++q) {
    float4 zv = zp[q], av = ap[q], rv = rp[q];
    sl += zv.x * av.x + zv.y * av.y + zv.z * av.z + zv.w * av.w;
    sr += zv.x * rv.x + zv.y * rv.y + zv.z * rv.z + zv.w * rv.w;
  }
  el[idx] = sl; er[idx] = sr;
}

// pass 1: e = leaky_relu(el[src]+er[dst]); segment-max via encoded atomicMax
__global__ void edge_pass1(const int* __restrict__ src, const int* __restrict__ dst,
                           const float* __restrict__ el, const float* __restrict__ er,
                           float* __restrict__ ebuf, unsigned* __restrict__ emax)
{
  long idx = (long)blockIdx.x * blockDim.x + threadIdx.x;   // e*H + h
  if (idx >= (long)EE * H) return;
  int e = (int)(idx >> 2), hh = (int)(idx & 3);
  int s = src[e], d = dst[e];
  float v = el[s * H + hh] + er[d * H + hh];
  v = (v > 0.f) ? v : NEG_SLOPE * v;
  ebuf[idx] = v;
  atomicMax(&emax[d * H + hh], encf(v));
}

// pass 2: ex = exp(e - max); denom[dst] += ex
__global__ void edge_pass2(const int* __restrict__ dst,
                           float* __restrict__ ebuf,
                           const unsigned* __restrict__ emax,
                           float* __restrict__ denom)
{
  long idx = (long)blockIdx.x * blockDim.x + threadIdx.x;
  if (idx >= (long)EE * H) return;
  int e = (int)(idx >> 2), hh = (int)(idx & 3);
  int d = dst[e];
  float m  = decf(emax[d * H + hh]);
  float ex = __expf(ebuf[idx] - m);
  ebuf[idx] = ex;
  atomicAdd(&denom[d * H + hh], ex);
}

// pass 3: out[dst,h,:] += (ex/denom[dst,h]) * z[src,h,:]
__global__ void edge_pass3(const int* __restrict__ src, const int* __restrict__ dst,
                           const float* __restrict__ ebuf,
                           const float* __restrict__ denom,
                           const float* __restrict__ z,
                           float* __restrict__ outemb)
{
  long idx = (long)blockIdx.x * blockDim.x + threadIdx.x;
  if (idx >= (long)EE * H) return;
  int e = (int)(idx >> 2), hh = (int)(idx & 3);
  // prefetch the index stream ahead (speculative; -> global_prefetch_b8)
  __builtin_prefetch(src + min(e + 4096, EE - 1), 0, 0);
  __builtin_prefetch(dst + min(e + 4096, EE - 1), 0, 0);
  int s = src[e], d = dst[e];
  float alpha = ebuf[idx] / denom[d * H + hh];
  const float4* zp = (const float4*)(z + (size_t)s * F + hh * D);
  float* op = outemb + (size_t)d * F + hh * D;
#pragma unroll
  for (int q = 0; q < D / 4; ++q) {
    float4 zv = zp[q];
    atomicAdd(op + q * 4 + 0, alpha * zv.x);
    atomicAdd(op + q * 4 + 1, alpha * zv.y);
    atomicAdd(op + q * 4 + 2, alpha * zv.z);
    atomicAdd(op + q * 4 + 3, alpha * zv.w);
  }
}

// emb = elu(emb + b)
__global__ void bias_elu(float* __restrict__ emb, const float* __restrict__ b)
{
  long idx = (long)blockIdx.x * blockDim.x + threadIdx.x;
  if (idx >= (long)NN * F) return;
  float v = emb[idx] + b[idx & (F - 1)];
  emb[idx] = (v > 0.f) ? v : (__expf(v) - 1.f);
}

// per-row dot(tanhbuf[r,:], w2) accumulated (block-partial) into acc[c]
__global__ void sa_reduce(const float* __restrict__ tb,
                          const float* __restrict__ w2,
                          float* __restrict__ acc, int Mc)
{
  __shared__ float part[4];
  if (threadIdx.x < 4) part[threadIdx.x] = 0.f;
  __syncthreads();
  long r = (long)blockIdx.x * blockDim.x + threadIdx.x;
  if (r < (long)Mc * NN) {
    const float4* tp = (const float4*)(tb + r * SAH);
    const float4* wp = (const float4*)w2;
    float sum = 0.f;
#pragma unroll
    for (int q = 0; q < SAH / 4; ++q) {
      float4 a = tp[q], b = wp[q];
      sum += a.x * b.x + a.y * b.y + a.z * b.z + a.w * b.w;
    }
    int c = (int)(r / NN);
    atomicAdd(&part[c], sum);
  }
  __syncthreads();
  if ((int)threadIdx.x < Mc) atomicAdd(&acc[threadIdx.x], part[threadIdx.x]);
}

__global__ void beta_softmax(const float* __restrict__ acc, float* __restrict__ beta, int Mc)
{
  float w[4];
  float m = -1e30f;
  for (int c = 0; c < Mc; ++c) { w[c] = acc[c] / (float)NN; m = fmaxf(m, w[c]); }
  float s = 0.f;
  for (int c = 0; c < Mc; ++c) { w[c] = __expf(w[c] - m); s += w[c]; }
  for (int c = 0; c < Mc; ++c) beta[c] = w[c] / s;
}

// out[n,f] = sum_c beta[c] * emb[c,n,f]
__global__ void combine(const float* __restrict__ emb, const float* __restrict__ beta,
                        float* __restrict__ outp, int Mc)
{
  long idx = (long)blockIdx.x * blockDim.x + threadIdx.x;
  if (idx >= (long)NN * F) return;
  float s = 0.f;
  for (int c = 0; c < Mc; ++c) s += beta[c] * emb[(size_t)c * NN * F + idx];
  outp[idx] = s;
}

// ---------------- host-side helpers ----------------
static inline void run_gemm(const unsigned short* A, int lda, const unsigned short* Bt,
                            float* C, int ldc, int M, int Nc, int K,
                            const float* bias, int act, hipStream_t s)
{
  dim3 grid((Nc + 63) / 64, (M + 63) / 64);
  if (bias) {
    if (act == 1) gemm_bf16_wmma<true, 1><<<grid, 256, 0, s>>>(A, lda, Bt, C, ldc, M, Nc, K, bias);
    else          gemm_bf16_wmma<true, 0><<<grid, 256, 0, s>>>(A, lda, Bt, C, ldc, M, Nc, K, bias);
  } else {
    gemm_bf16_wmma<false, 0><<<grid, 256, 0, s>>>(A, lda, Bt, C, ldc, M, Nc, K, nullptr);
  }
}

static inline void run_transpose(const float* B, unsigned short* Bt, int K, int Nc, int ldb,
                                 hipStream_t s)
{
  transpose_bf16<<<(K * Nc + 255) / 256, 256, 0, s>>>(B, Bt, K, Nc, ldb);
}

static inline void run_cvt(const float* in, unsigned short* out, long n, hipStream_t s)
{
  cvt_bf16<<<(unsigned)((n + 255) / 256), 256, 0, s>>>(in, out, n);
}

static inline void run_conv(const unsigned short* x_bf, const float* W, const float* al,
                            const float* ar, const float* bvec,
                            const int* src, const int* dst,
                            unsigned short* bt, float* zproj, float* embslice,
                            float* el, float* er, unsigned* emax, float* denom,
                            float* ebuf, hipStream_t s)
{
  // z = x @ W   [NN,64] x [64,256]  (z stays L2-resident: 51 MB of 192 MB)
  run_transpose(W, bt, HID, F, F, s);
  run_gemm(x_bf, HID, bt, zproj, F, NN, F, HID, nullptr, 0, s);
  attn_scores<<<(NN * H + 255) / 256, 256, 0, s>>>(zproj, al, ar, el, er);
  fill_u32<<<(NN * H + 255) / 256, 256, 0, s>>>(emax, 0x007FFFFFu /*enc(-inf)*/, NN * H);
  zero_f32<<<(NN * H + 255) / 256, 256, 0, s>>>(denom, NN * H);
  zero_f32<<<((long)NN * F + 255) / 256, 256, 0, s>>>(embslice, (long)NN * F);
  long eh = (long)EE * H;
  edge_pass1<<<(unsigned)((eh + 255) / 256), 256, 0, s>>>(src, dst, el, er, ebuf, emax);
  edge_pass2<<<(unsigned)((eh + 255) / 256), 256, 0, s>>>(dst, ebuf, emax, denom);
  edge_pass3<<<(unsigned)((eh + 255) / 256), 256, 0, s>>>(src, dst, ebuf, denom, zproj, embslice);
  bias_elu<<<(unsigned)(((long)NN * F + 255) / 256), 256, 0, s>>>(embslice, bvec);
}

static inline void run_sematt(const float* embbase, int Mc,
                              const float* W1, const float* b1, const float* w2,
                              unsigned short* abuf, unsigned short* bt,
                              float* tanhbuf, float* acc, float* beta,
                              float* outp, hipStream_t s)
{
  // tanh(z @ W1 + b1) with fused bias+tanh WMMA epilogue
  run_cvt(embbase, abuf, (long)Mc * NN * F, s);
  run_transpose(W1, bt, F, SAH, SAH, s);
  run_gemm(abuf, F, bt, tanhbuf, SAH, Mc * NN, SAH, F, b1, 1, s);
  zero_f32<<<1, 32, 0, s>>>(acc, 8);
  sa_reduce<<<(unsigned)(((long)Mc * NN + 255) / 256), 256, 0, s>>>(tanhbuf, w2, acc, Mc);
  beta_softmax<<<1, 1, 0, s>>>(acc, beta, Mc);
  combine<<<(unsigned)(((long)NN * F + 255) / 256), 256, 0, s>>>(embbase, beta, outp, Mc);
}

extern "C" void kernel_launch(void* const* d_in, const int* in_sizes, int n_in,
                              void* d_out, int out_size, void* d_ws, size_t ws_size,
                              hipStream_t stream)
{
  (void)in_sizes; (void)n_in; (void)out_size; (void)ws_size;

  const float* h_in  = (const float*)d_in[0];
  const int* src[4]  = {(const int*)d_in[1], (const int*)d_in[3],
                        (const int*)d_in[5], (const int*)d_in[7]};
  const int* dst[4]  = {(const int*)d_in[2], (const int*)d_in[4],
                        (const int*)d_in[6], (const int*)d_in[8]};
  const float* fcW   = (const float*)d_in[10];
  const float* W123  = (const float*)d_in[11];
  const float* al123 = (const float*)d_in[12];
  const float* ar123 = (const float*)d_in[13];
  const float* b123  = (const float*)d_in[14];
  const float* saW1_123 = (const float*)d_in[15];
  const float* sab1_123 = (const float*)d_in[16];
  const float* saw2_123 = (const float*)d_in[17];
  const float* W124  = (const float*)d_in[18];
  const float* al124 = (const float*)d_in[19];
  const float* ar124 = (const float*)d_in[20];
  const float* b124  = (const float*)d_in[21];
  const float* saW1_124 = (const float*)d_in[22];
  const float* sab1_124 = (const float*)d_in[23];
  const float* saw2_124 = (const float*)d_in[24];
  const float* saW1_fin = (const float*)d_in[25];
  const float* sab1_fin = (const float*)d_in[26];
  const float* saw2_fin = (const float*)d_in[27];
  const float* predW = (const float*)d_in[28];
  const float* predb = (const float*)d_in[29];

  // ---- workspace carve: f32 region then bf16 region (~510 MB total, heavy reuse)
  float* base = (float*)d_ws;
  size_t o = 0;
  float* x       = base + o; o += (size_t)NN * HID;      // fc output [N,64]
  float* zproj   = base + o; o += (size_t)NN * F;        // per-conv z (reused)
  float* emb     = base + o; o += (size_t)3 * NN * F;    // per-layer [3,N,256] (reused)
  float* hl      = base + o; o += (size_t)2 * NN * F;    // h_123 / h_124 stacked
  float* tanhbuf = base + o; o += (size_t)3 * NN * SAH;  // reused
  float* ebuf    = base + o; o += (size_t)EE * H;        // per-conv edge buffer (reused)
  float* el      = base + o; o += (size_t)NN * H;
  float* er      = base + o; o += (size_t)NN * H;
  unsigned* emax = (unsigned*)(base + o); o += (size_t)NN * H;
  float* denom   = base + o; o += (size_t)NN * H;
  float* acc     = base + o; o += 8;
  float* beta    = base + o; o += 8;
  o = (o + 3) & ~(size_t)3;                              // 16B-align bf16 region
  unsigned short* h_bf = (unsigned short*)(base + o); o += (size_t)NN * INF_ / 2;
  unsigned short* x_bf = (unsigned short*)(base + o); o += (size_t)NN * HID / 2;
  unsigned short* abuf = (unsigned short*)(base + o); o += (size_t)3 * NN * F / 2; // A scratch (bf16)
  unsigned short* bt   = (unsigned short*)(base + o); o += 32768 / 2;              // Bt scratch (bf16)

  float* out_f  = (float*)d_out;
  float* hfinal = out_f + (size_t)NN * OUTF;   // h_final written straight into output

  // 1) x = h @ fc_W    [50000,128] x [128,64]
  run_cvt(h_in, h_bf, (long)NN * INF_, stream);
  run_transpose(fcW, bt, INF_, HID, HID, stream);
  run_gemm(h_bf, INF_, bt, x, HID, NN, HID, INF_, nullptr, 0, stream);
  run_cvt(x, x_bf, (long)NN * HID, stream);   // A operand for all 6 convs

  // 2) layer 123: graphs {0,1,2}
  {
    const int g[3] = {0, 1, 2};
    for (int i = 0; i < 3; ++i)
      run_conv(x_bf, W123 + (size_t)i * HID * F, al123 + (size_t)i * F,
               ar123 + (size_t)i * F, b123 + (size_t)i * F,
               src[g[i]], dst[g[i]], bt, zproj, emb + (size_t)i * NN * F,
               el, er, emax, denom, ebuf, stream);
    run_sematt(emb, 3, saW1_123, sab1_123, saw2_123, abuf, bt,
               tanhbuf, acc, beta, hl, stream);
  }

  // 3) layer 124: graphs {0,1,3}
  {
    const int g[3] = {0, 1, 3};
    for (int i = 0; i < 3; ++i)
      run_conv(x_bf, W124 + (size_t)i * HID * F, al124 + (size_t)i * F,
               ar124 + (size_t)i * F, b124 + (size_t)i * F,
               src[g[i]], dst[g[i]], bt, zproj, emb + (size_t)i * NN * F,
               el, er, emax, denom, ebuf, stream);
    run_sematt(emb, 3, saW1_124, sab1_124, saw2_124, abuf, bt,
               tanhbuf, acc, beta, hl + (size_t)NN * F, stream);
  }

  // 4) final semantic attention over {h_123, h_124} -> h_final (into d_out)
  run_sematt(hl, 2, saW1_fin, sab1_fin, saw2_fin, abuf, bt,
             tanhbuf, acc, beta, hfinal, stream);

  // 5) pred = h_final @ pred_W + pred_b   [50000,256] x [256,16] (into d_out)
  run_cvt(hfinal, abuf, (long)NN * F, stream);
  run_transpose(predW, bt, F, OUTF, OUTF, stream);
  run_gemm(abuf, F, bt, out_f, OUTF, NN, OUTF, F, predb, 0, stream);
}
